// fbGSC_18296560681669
// MI455X (gfx1250) — matrix operand order repaired and checked by reading, not compiled, not run on hardware
//
#include <hip/hip_runtime.h>
#include <stdint.h>

// ---------------- problem constants ----------------
#define T_STEPS 101
#define BATCH   16384
#define IN_DIM  40
#define HID     300
#define OUT_DIM 12

// ---------------- tiling ----------------
#define NT 19                 // 19 * 16 = 304 >= 300   (N tiles)
#define KC 10                 // 10 * 32 = 320 >= 300   (K chunks)
#define WS 320                // padded K stride (bf16 elements)
#define W1KS 64               // padded K stride for W1 (40 -> 64)
#define WAVES 4
#define MROWS 32              // batch rows per wave (two 16-row A tiles)
#define ROWS_PER_WG (WAVES * MROWS)       // 128
#define NWG (BATCH / ROWS_PER_WG)         // 128 workgroups

typedef __attribute__((ext_vector_type(16))) __bf16        v16bf;
typedef __attribute__((ext_vector_type(8)))  float         v8f;
typedef __attribute__((ext_vector_type(8)))  unsigned int  v8u;

static __device__ __forceinline__ unsigned short f2bf(float f){
  unsigned int u = __builtin_bit_cast(unsigned int, f);
  u += 0x7FFFu + ((u >> 16) & 1u);            // round-to-nearest-even
  return (unsigned short)(u >> 16);
}
static __device__ __forceinline__ v8f zero8(){
  v8f z;
#pragma unroll
  for (int i = 0; i < 8; ++i) z[i] = 0.0f;
  return z;
}
// B fragment (32x16 bf16): per-lane two contiguous 16B K-slices (kb and kb+16) of row-major W (B = W^T)
static __device__ __forceinline__ v16bf ld_fragB(const unsigned short* p){
  uint4 lo = *(const uint4*)p;
  uint4 hi = *(const uint4*)(p + 16);
  v8u u = { lo.x, lo.y, lo.z, lo.w, hi.x, hi.y, hi.z, hi.w };
  return __builtin_bit_cast(v16bf, u);
}
// A fragment (16x32 bf16): per-lane two contiguous 16B K-slices (kb and kb+16)
static __device__ __forceinline__ v16bf ld_fragA(const unsigned short* p){
  uint4 lo = *(const uint4*)p;
  uint4 hi = *(const uint4*)(p + 16);
  v8u u = { lo.x, lo.y, lo.z, lo.w, hi.x, hi.y, hi.z, hi.w };
  return __builtin_bit_cast(v16bf, u);
}
static __device__ __forceinline__ v16bf mk_frag(const unsigned short e[16]){
  v8u u;
#pragma unroll
  for (int v = 0; v < 8; ++v)
    u[v] = (unsigned int)e[2*v] | ((unsigned int)e[2*v+1] << 16);
  return __builtin_bit_cast(v16bf, u);
}
static __device__ __forceinline__ v16bf frag_from_bits(unsigned int bw, int kb8){
  unsigned short e[16];
#pragma unroll
  for (int j = 0; j < 8; ++j){
    e[j]     = ((bw >> (kb8 + j))      & 1u) ? (unsigned short)0x3F80 : (unsigned short)0;
    e[8 + j] = ((bw >> (kb8 + 16 + j)) & 1u) ? (unsigned short)0x3F80 : (unsigned short)0;
  }
  return mk_frag(e);
}
static __device__ __forceinline__ v8f wmma_bf16(v16bf a, v16bf b, v8f c){
  return __builtin_amdgcn_wmma_f32_16x16x32_bf16(false, a, false, b,
                                                 (short)0, c, false, false);
}
// build the two x_t A-fragments (K padded 40 -> 64) for one 16-row tile
static __device__ __forceinline__ void mk_xfrags(const float* xr, int kb8,
                                                 v16bf& x0, v16bf& x1){
  unsigned short e[16];
#pragma unroll
  for (int j = 0; j < 8; ++j){ e[j] = f2bf(xr[kb8 + j]); e[8 + j] = f2bf(xr[16 + kb8 + j]); }
  x0 = mk_frag(e);
#pragma unroll
  for (int j = 0; j < 8; ++j){
    int k = 32 + kb8 + j;
    e[j] = (k < IN_DIM) ? f2bf(xr[k]) : (unsigned short)0;
    e[8 + j] = 0;
  }
  x1 = mk_frag(e);
}

__global__ __launch_bounds__(WAVES * 32, 1)
void snn_lif_wmma_kernel(const float* __restrict__ x,  const float* __restrict__ W1,
                         const float* __restrict__ b1, const float* __restrict__ Wh,
                         const float* __restrict__ bh, const float* __restrict__ W2,
                         const float* __restrict__ b2, const float* __restrict__ W3,
                         const float* __restrict__ b3, float* __restrict__ out,
                         unsigned int* __restrict__ ybits)
{
  // LDS: big weight (Wh then W2), small weight (W1 then W3), bias, spike tiles
  __shared__ __align__(32) unsigned short sW   [304 * WS];        // 194560 B
  __shared__ __align__(32) unsigned short sWS  [304 * W1KS];      //  38912 B
  __shared__                float          sBias[304];             //   1216 B
  __shared__ __align__(32) unsigned short sTile[WAVES * MROWS * WS]; // 81920 B
  // total 316608 B  (<= 320 KB per workgroup)

  const int tid  = threadIdx.x;
  const int lane = tid & 31;
  const int wv   = tid >> 5;
  const int b0w  = blockIdx.x * ROWS_PER_WG + wv * MROWS; // this wave's 32 batch rows

  const int am   = lane & 15;          // A row in tile / B column in tile
  const int kb8  = (lane >> 4) * 8;    // A-fragment K sub-offset
  const int kb16 = (lane >> 4) * 8;    // B-fragment K sub-offset
  unsigned short* myTile = sTile + wv * MROWS * WS;

  // ================= Phase A setup: Wh, W1, bh+b1 -> LDS (bf16, zero-padded)
  for (int idx = tid; idx < 304 * WS; idx += WAVES * 32){
    int n = idx / WS, k = idx - n * WS;
    sW[idx] = f2bf((n < HID && k < HID) ? Wh[n * HID + k] : 0.0f);
  }
  for (int idx = tid; idx < 304 * W1KS; idx += WAVES * 32){
    int n = idx / W1KS, k = idx - n * W1KS;
    sWS[idx] = f2bf((n < HID && k < IN_DIM) ? W1[n * IN_DIM + k] : 0.0f);
  }
  for (int idx = tid; idx < 304; idx += WAVES * 32)
    sBias[idx] = (idx < HID) ? (bh[idx] + b1[idx]) : 0.0f;
  for (int idx = tid; idx < WAVES * MROWS * WS; idx += WAVES * 32)
    sTile[idx] = 0;                                   // y(−1) = 0, K-pad = 0
  __syncthreads();

  // membrane potential v1 in registers, D-layout, two 16-row halves
  v8f v1a[NT], v1b[NT];
#pragma unroll
  for (int i = 0; i < NT; ++i){ v1a[i] = zero8(); v1b[i] = zero8(); }

  // ================= Phase A: recurrent layer, emit bit-packed spike stream
  for (int t = 0; t < T_STEPS; ++t){
    asm volatile("s_wait_dscnt 0" ::: "memory");      // prev-step spike stores visible
    // y(t-1) A-fragments for both row halves
    v16bf yA0[KC], yA1[KC];
#pragma unroll
    for (int kc = 0; kc < KC; ++kc){
      yA0[kc] = ld_fragA(&myTile[ am       * WS + kc * 32 + kb8]);
      yA1[kc] = ld_fragA(&myTile[(16 + am) * WS + kc * 32 + kb8]);
    }
    // x_t A-fragments (f32 -> bf16 on the fly)
    const float* xr0 = x + ((size_t)t * BATCH + b0w + am) * IN_DIM;
    const float* xr1 = xr0 + 16 * IN_DIM;
    __builtin_prefetch(xr0 + (size_t)BATCH * IN_DIM, 0, 1);  // next step (speculative)
    v16bf xA00, xA01, xA10, xA11;
    mk_xfrags(xr0, kb8, xA00, xA01);
    mk_xfrags(xr1, kb8, xA10, xA11);

    unsigned int yb0[KC], yb1[KC];
#pragma unroll
    for (int d = 0; d < KC; ++d){ yb0[d] = 0; yb1[d] = 0; }

#pragma unroll
    for (int nt = 0; nt < NT; ++nt){
      v8f acc0 = zero8(), acc1 = zero8();
      // h1_t contribution: x_t @ W1^T (2 K-chunks); each B fragment feeds 2 WMMAs
      {
        v16bf bf = ld_fragB(&sWS[(nt * 16 + am) * W1KS + 0 + kb16]);
        acc0 = wmma_bf16(xA00, bf, acc0);
        acc1 = wmma_bf16(xA10, bf, acc1);
        bf = ld_fragB(&sWS[(nt * 16 + am) * W1KS + 32 + kb16]);
        acc0 = wmma_bf16(xA01, bf, acc0);
        acc1 = wmma_bf16(xA11, bf, acc1);
      }
      // recurrent: y(t-1) @ Wh^T (10 K-chunks), B fragment reused across halves
#pragma unroll
      for (int kc = 0; kc < KC; ++kc){
        v16bf bf = ld_fragB(&sW[(nt * 16 + am) * WS + kc * 32 + kb16]);
        acc0 = wmma_bf16(yA0[kc], bf, acc0);
        acc1 = wmma_bf16(yA1[kc], bf, acc1);
      }
      float bb = sBias[nt * 16 + am];
#pragma unroll
      for (int i = 0; i < 8; ++i){
        int m = (lane < 16) ? i : (i + 8);           // D-layout row within tile
        // half 0
        {
          float cur = acc0[i] + bb;
          float vv  = 0.5f * (v1a[nt][i] + cur);     // v += (cur - v)/tau, tau = 2
          bool  sp  = (vv >= 1.0f);                  // Heaviside(v - vth)
          v1a[nt][i] = sp ? 0.0f : vv;               // hard reset
          myTile[m * WS + nt * 16 + am] = sp ? (unsigned short)0x3F80 : (unsigned short)0;
          unsigned int bm = __builtin_amdgcn_ballot_w32(sp);   // wave32
          if (am == i){
            unsigned int h = (lane < 16) ? (bm & 0xFFFFu) : (bm >> 16);
            yb0[nt >> 1] |= h << ((nt & 1) * 16);
          }
        }
        // half 1
        {
          float cur = acc1[i] + bb;
          float vv  = 0.5f * (v1b[nt][i] + cur);
          bool  sp  = (vv >= 1.0f);
          v1b[nt][i] = sp ? 0.0f : vv;
          myTile[(16 + m) * WS + nt * 16 + am] = sp ? (unsigned short)0x3F80 : (unsigned short)0;
          unsigned int bm = __builtin_amdgcn_ballot_w32(sp);
          if (am == i){
            unsigned int h = (lane < 16) ? (bm & 0xFFFFu) : (bm >> 16);
            yb1[nt >> 1] |= h << ((nt & 1) * 16);
          }
        }
      }
    }
    // bit-packed y(t) -> workspace (66 MB total, L2-resident)
    if (am < 8){
      int m = (lane < 16) ? am : (am + 8);
      unsigned int* dst0 = ybits + ((size_t)t * BATCH + b0w + m) * KC;
      unsigned int* dst1 = ybits + ((size_t)t * BATCH + b0w + 16 + m) * KC;
#pragma unroll
      for (int d = 0; d < KC; ++d){ dst0[d] = yb0[d]; dst1[d] = yb1[d]; }
    }
  }

  // ================= Phase B setup: W2, W3, b2 -> LDS
  __syncthreads();
  for (int idx = tid; idx < 304 * WS; idx += WAVES * 32){
    int n = idx / WS, k = idx - n * WS;
    sW[idx] = f2bf((n < HID && k < HID) ? W2[n * HID + k] : 0.0f);
  }
  for (int idx = tid; idx < 16 * WS; idx += WAVES * 32){   // W3 reuses sWS region
    int n = idx / WS, k = idx - n * WS;
    sWS[idx] = f2bf((n < OUT_DIM && k < HID) ? W3[n * HID + k] : 0.0f);
  }
  for (int idx = tid; idx < 304; idx += WAVES * 32)
    sBias[idx] = (idx < HID) ? b2[idx] : 0.0f;
  for (int idx = tid; idx < WAVES * MROWS * WS; idx += WAVES * 32)
    sTile[idx] = 0;
  __syncthreads();

  v8f v2a[NT], v2b[NT];
#pragma unroll
  for (int i = 0; i < NT; ++i){ v2a[i] = zero8(); v2b[i] = zero8(); }
  v8f oacc0 = zero8(), oacc1 = zero8();

  // ================= Phase B: feedforward LIF + output accumulation
  for (int t = 0; t < T_STEPS; ++t){
    // unpack y(t) bits -> bf16 A-fragments for both halves
    unsigned int w0[KC], w1[KC];
    const unsigned int* src0 = ybits + ((size_t)t * BATCH + b0w + am) * KC;
    const unsigned int* src1 = src0 + 16 * KC;
#pragma unroll
    for (int d = 0; d < KC; ++d){ w0[d] = src0[d]; w1[d] = src1[d]; }
    v16bf yA0[KC], yA1[KC];
#pragma unroll
    for (int kc = 0; kc < KC; ++kc){
      yA0[kc] = frag_from_bits(w0[kc], kb8);
      yA1[kc] = frag_from_bits(w1[kc], kb8);
    }

#pragma unroll
    for (int nt = 0; nt < NT; ++nt){
      v8f acc0 = zero8(), acc1 = zero8();
#pragma unroll
      for (int kc = 0; kc < KC; ++kc){
        v16bf bf = ld_fragB(&sW[(nt * 16 + am) * WS + kc * 32 + kb16]);
        acc0 = wmma_bf16(yA0[kc], bf, acc0);
        acc1 = wmma_bf16(yA1[kc], bf, acc1);
      }
      float bb = sBias[nt * 16 + am];
#pragma unroll
      for (int i = 0; i < 8; ++i){
        int m = (lane < 16) ? i : (i + 8);
        {
          float cur = acc0[i] + bb;
          float vv  = 0.5f * (v2a[nt][i] + cur);
          bool  sp  = (vv >= 1.0f);
          v2a[nt][i] = sp ? 0.0f : vv;
          myTile[m * WS + nt * 16 + am] = sp ? (unsigned short)0x3F80 : (unsigned short)0;
        }
        {
          float cur = acc1[i] + bb;
          float vv  = 0.5f * (v2b[nt][i] + cur);
          bool  sp  = (vv >= 1.0f);
          v2b[nt][i] = sp ? 0.0f : vv;
          myTile[(16 + m) * WS + nt * 16 + am] = sp ? (unsigned short)0x3F80 : (unsigned short)0;
        }
      }
    }
    asm volatile("s_wait_dscnt 0" ::: "memory");      // s2 tile visible to whole wave
    // out += s2 @ W3^T  (single N tile, 10 K-chunks, B fragment shared by halves)
#pragma unroll
    for (int kc = 0; kc < KC; ++kc){
      v16bf bf  = ld_fragB(&sWS[am * WS + kc * 32 + kb16]);
      v16bf sa0 = ld_fragA(&myTile[ am       * WS + kc * 32 + kb8]);
      v16bf sa1 = ld_fragA(&myTile[(16 + am) * WS + kc * 32 + kb8]);
      oacc0 = wmma_bf16(sa0, bf, oacc0);
      oacc1 = wmma_bf16(sa1, bf, oacc1);
    }
  }

  // ================= epilogue: out = sum_t s2@W3^T + T*b3
  float b3v = (am < OUT_DIM) ? b3[am] : 0.0f;
#pragma unroll
  for (int i = 0; i < 8; ++i){
    int m = (lane < 16) ? i : (i + 8);
    if (am < OUT_DIM){
      out[(size_t)(b0w + m)      * OUT_DIM + am] = oacc0[i] + (float)T_STEPS * b3v;
      out[(size_t)(b0w + 16 + m) * OUT_DIM + am] = oacc1[i] + (float)T_STEPS * b3v;
    }
  }
}

extern "C" void kernel_launch(void* const* d_in, const int* in_sizes, int n_in,
                              void* d_out, int out_size, void* d_ws, size_t ws_size,
                              hipStream_t stream)
{
  (void)in_sizes; (void)n_in; (void)out_size; (void)ws_size;
  const float* x  = (const float*)d_in[0];
  const float* W1 = (const float*)d_in[1];
  const float* b1 = (const float*)d_in[2];
  const float* Wh = (const float*)d_in[3];
  const float* bh = (const float*)d_in[4];
  const float* W2 = (const float*)d_in[5];
  const float* b2 = (const float*)d_in[6];
  const float* W3 = (const float*)d_in[7];
  const float* b3 = (const float*)d_in[8];
  // workspace: bit-packed spike stream, T*B*KC u32 = 66.2 MB (fits 192 MB L2)
  snn_lif_wmma_kernel<<<NWG, WAVES * 32, 0, stream>>>(
      x, W1, b1, Wh, bh, W2, b2, W3, b3, (float*)d_out, (unsigned int*)d_ws);
}
